// MambaProjectorV2_84293028151646
// MI455X (gfx1250) — compile-verified, matrix operand order checked by reading, not run on hardware
//
#include <hip/hip_runtime.h>
#include <stdint.h>

// ---------------- problem constants (from the reference) ----------------
#define Bc     4
#define Lc     576
#define FUSEDc 2048
#define DIMc   2048
#define DIc    4096     // 2*DIM
#define DSc    16
#define DCONVc 4
#define DTRc   128
#define Mrows  (Bc * Lc)          // 2304 = 18 * 128 (M-tile divides evenly)
#define XPN    (DTRc + 2 * DSc)   // 160
#define LROW   40                 // LDS row stride in ushorts (64B data + 16B pad)

typedef unsigned short ushort_t;
typedef __attribute__((ext_vector_type(16))) __bf16 bf16x16;
typedef __attribute__((ext_vector_type(8)))  float  floatx8;
typedef __attribute__((ext_vector_type(4)))  unsigned int u32x4;
typedef __attribute__((ext_vector_type(8)))  int          i32x8;
typedef __attribute__((ext_vector_type(4)))  int          i32x4;

// ---------------- bf16 <-> f32 helpers (RNE) ----------------
__device__ __forceinline__ float bf2f(ushort_t u) {
  union { unsigned int i; float f; } c;
  c.i = ((unsigned int)u) << 16;
  return c.f;
}
__device__ __forceinline__ ushort_t f2bf(float f) {
  union { float f; unsigned int i; } c;
  c.f = f;
  unsigned int u = c.i;
  u += 0x7FFFu + ((u >> 16) & 1u);   // round to nearest even
  return (ushort_t)(u >> 16);
}

__device__ __forceinline__ float apply_act(float v, int act) {
  if (act == 1) {            // exact GELU
    return 0.5f * v * (1.0f + erff(v * 0.70710678118654752f));
  } else if (act == 2) {     // softplus
    return (v > 20.0f) ? v : log1pf(__expf(v));
  }
  return v;
}

// ---------------- TDM: async 2D tile (rows x 32 bf16) global -> LDS -------
// Pads LDS dest 16B after every 64B row => 80B LDS row stride (bank spread).
// OOB rows (beyond rowsRemain) are zero-filled by the TDM.
__device__ __forceinline__ void tdm_load_tile(unsigned ldsOff,
                                              const ushort_t* gptr,
                                              int dim0Remain, int rowsRemain,
                                              int lda) {
  unsigned long long ga = (unsigned long long)(size_t)gptr;
  u32x4 g0;
  g0[0] = 1u;                                    // count=1, user mode
  g0[1] = ldsOff;                                // lds_addr
  g0[2] = (unsigned)ga;                          // global_addr[31:0]
  g0[3] = (unsigned)((ga >> 32) & 0x01FFFFFFu)   // global_addr[56:32]
          | (2u << 30);                          // type = 2 (image)
  unsigned d0 = (unsigned)dim0Remain;
  unsigned d1 = (unsigned)(rowsRemain < 0 ? 0 : rowsRemain);
  i32x8 g1;
  g1[0] = (int)((1u << 16)        // data_size = 2 bytes
                | (1u << 20)      // pad_enable
                | (3u << 22)      // pad_interval: 16 DWORDs (64B)
                | (3u << 25));    // pad_amount: 4 DWORDs (16B)
  g1[1] = (int)((d0 & 0xFFFFu) << 16);                       // tensor_dim0 lo
  g1[2] = (int)(((d0 >> 16) & 0xFFFFu) | ((d1 & 0xFFFFu) << 16));
  g1[3] = (int)(((d1 >> 16) & 0xFFFFu) | (32u << 16));       // tile_dim0 = 32
  g1[4] = (int)128;                                          // tile_dim1 = 128
  g1[5] = (int)(unsigned)lda;                                // dim0_stride lo32
  g1[6] = 0;
  g1[7] = 0;
  i32x4 gz4 = {0, 0, 0, 0};
  i32x8 gz8 = {0, 0, 0, 0, 0, 0, 0, 0};
  // 6-arg toolchain variant: (g0, g1, g2, g3, g4, cpol)
  __builtin_amdgcn_tensor_load_to_lds(g0, g1, gz4, gz4, gz8, 0);
}

// ---------------- fp32 -> bf16 convert ----------------
__global__ void f32_to_bf16_kernel(const float* __restrict__ src,
                                   ushort_t* __restrict__ dst, size_t n) {
  for (size_t i = blockIdx.x * (size_t)blockDim.x + threadIdx.x; i < n;
       i += (size_t)gridDim.x * blockDim.x)
    dst[i] = f2bf(src[i]);
}

// ---------------- flip along L (bf16, (B,L,DIM)) ----------------
__global__ void flip_bf16_kernel(const ushort_t* __restrict__ src,
                                 ushort_t* __restrict__ dst) {
  size_t total = (size_t)Mrows * DIMc;
  for (size_t idx = blockIdx.x * (size_t)blockDim.x + threadIdx.x; idx < total;
       idx += (size_t)gridDim.x * blockDim.x) {
    int d = (int)(idx % DIMc);
    int t = (int)(idx / DIMc);
    int l = t % Lc, b = t / Lc;
    dst[idx] = src[((size_t)b * Lc + (Lc - 1 - l)) * DIMc + d];
  }
}

// ---------------- WMMA bf16 GEMM: C(M,N) = act(A(M,K) * W(N,K)^T + bias) ---
// Block tile 128(M) x 128(N); 8 waves as 4(M) x 2(N); wave tile 32x64 = 2x4
// wmma fragments. K-step 32, TDM double-buffered global->LDS staging.
// M must be a multiple of 128 (all calls use M = 2304). N is guarded.
union FragU { bf16x16 v; uint4 q[2]; };

__global__ void __launch_bounds__(256)
gemm_bf16_wmma(const ushort_t* __restrict__ A, int lda,
               const ushort_t* __restrict__ W, int ldb,
               const float* __restrict__ bias,
               float* __restrict__ outF, ushort_t* __restrict__ outH, int ldo,
               int M, int N, int K, int act) {
  __shared__ ushort_t sA[2][128 * LROW];   // 2 x 10 KB
  __shared__ ushort_t sB[2][128 * LROW];   // 2 x 10 KB

  const int tid  = threadIdx.x;
  const int lane = tid & 31;
  const int wid  = tid >> 5;
  const int lr   = lane & 15;
  const int kh   = lane >> 4;              // K-half select per ISA layout
  const int waveM = (wid & 3) * 32;
  const int waveN = (wid >> 2) * 64;
  const int blockM = blockIdx.y * 128;
  const int blockN = blockIdx.x * 128;

  floatx8 zeroAcc = {};
  floatx8 acc[2][4];
#pragma unroll
  for (int mi = 0; mi < 2; ++mi)
#pragma unroll
    for (int ni = 0; ni < 4; ++ni) acc[mi][ni] = zeroAcc;

  const ushort_t* gA = A + (size_t)blockM * lda;
  const ushort_t* gW = W + (size_t)blockN * ldb;
  const int rowsA = M - blockM;            // >= 128 always
  const int rowsB = N - blockN;            // may be < 128 (TDM zero-fills)
  const int nK = K / 32;

  if (wid == 0) {                          // prologue: stage K-tile 0
    tdm_load_tile((unsigned)(size_t)&sA[0][0], gA, K, rowsA, lda);
    tdm_load_tile((unsigned)(size_t)&sB[0][0], gW, K, rowsB, ldb);
  }

  for (int i = 0; i < nK; ++i) {
    const int buf = i & 1;
    if (wid == 0) {
      if (i + 1 < nK) {                    // stage next tile, then make sure
        int k0 = (i + 1) * 32;             // the current one has landed
        tdm_load_tile((unsigned)(size_t)&sA[buf ^ 1][0], gA + k0, K - k0,
                      rowsA, lda);
        tdm_load_tile((unsigned)(size_t)&sB[buf ^ 1][0], gW + k0, K - k0,
                      rowsB, ldb);
        __builtin_amdgcn_s_wait_tensorcnt(2);
      } else {
        __builtin_amdgcn_s_wait_tensorcnt(0);
      }
    }
    __syncthreads();                       // current buffer visible to all

    const ushort_t* bufA = sA[buf];
    const ushort_t* bufB = sB[buf];
    // fragment loads per ISA 16-bit layout:
    // elems 0..7 -> K = kh*8 + 0..7 ; elems 8..15 -> K = 16 + kh*8 + 0..7
    FragU af[2], bfr[4];
#pragma unroll
    for (int mi = 0; mi < 2; ++mi) {
      int r = waveM + mi * 16 + lr;
      af[mi].q[0] = *(const uint4*)(&bufA[r * LROW + kh * 8]);
      af[mi].q[1] = *(const uint4*)(&bufA[r * LROW + 16 + kh * 8]);
    }
#pragma unroll
    for (int ni = 0; ni < 4; ++ni) {
      int c = waveN + ni * 16 + lr;
      bfr[ni].q[0] = *(const uint4*)(&bufB[c * LROW + kh * 8]);
      bfr[ni].q[1] = *(const uint4*)(&bufB[c * LROW + 16 + kh * 8]);
    }
#pragma unroll
    for (int mi = 0; mi < 2; ++mi)
#pragma unroll
      for (int ni = 0; ni < 4; ++ni)
        acc[mi][ni] = __builtin_amdgcn_wmma_f32_16x16x32_bf16(
            false, af[mi].v, false, bfr[ni].v, (short)0, acc[mi][ni],
            false, false);
    __syncthreads();                       // done reading buf before reuse
  }

  // epilogue: C/D layout -> VGPR i holds M=i (+8 for lanes 16..31), N=lane&15
#pragma unroll
  for (int mi = 0; mi < 2; ++mi) {
#pragma unroll
    for (int ni = 0; ni < 4; ++ni) {
      int n = blockN + waveN + ni * 16 + lr;
      if (n >= N) continue;
      float bv = bias ? bias[n] : 0.0f;
      int m0 = blockM + waveM + mi * 16 + (kh ? 8 : 0);
#pragma unroll
      for (int i = 0; i < 8; ++i) {
        float v = apply_act(acc[mi][ni][i] + bv, act);
        size_t o = (size_t)(m0 + i) * ldo + n;
        if (outF) outF[o] = v;
        if (outH) outH[o] = f2bf(v);
      }
    }
  }
}

// ---------------- causal depthwise conv (k=4) + SiLU ----------------
__global__ void conv_silu_kernel(const ushort_t* __restrict__ xz,
                                 const float* __restrict__ cw,
                                 const float* __restrict__ cb,
                                 ushort_t* __restrict__ xo) {
  size_t total = (size_t)Mrows * DIc;
  for (size_t idx = blockIdx.x * (size_t)blockDim.x + threadIdx.x; idx < total;
       idx += (size_t)gridDim.x * blockDim.x) {
    int d = (int)(idx % DIc);
    int t = (int)(idx / DIc);
    int l = t % Lc, b = t / Lc;
    float acc = cb[d];
#pragma unroll
    for (int kk = 0; kk < DCONVc; ++kk) {
      int ls = l + kk - (DCONVc - 1);
      if (ls >= 0)
        acc += cw[d * DCONVc + kk] *
               bf2f(xz[((size_t)(b * Lc + ls)) * (2 * DIc) + d]);
    }
    acc = acc / (1.0f + __expf(-acc));   // SiLU
    xo[idx] = f2bf(acc);
  }
}

// ---------------- selective scan: one thread per (b, d) channel ----------
__global__ void scan_kernel(const ushort_t* __restrict__ xi,
                            const float* __restrict__ dt,
                            const float* __restrict__ xdb,
                            const float* __restrict__ A_log,
                            const float* __restrict__ Dv,
                            float* __restrict__ y) {
  int idx = blockIdx.x * blockDim.x + threadIdx.x;
  if (idx >= Bc * DIc) return;
  int b = idx / DIc, d = idx % DIc;
  float Ar[DSc], hs[DSc];
#pragma unroll
  for (int s = 0; s < DSc; ++s) {
    Ar[s] = -__expf(A_log[d * DSc + s]);
    hs[s] = 0.0f;
  }
  float Dd = Dv[d];
  for (int l = 0; l < Lc; ++l) {
    size_t t = (size_t)b * Lc + l;
    float dtv = dt[t * DIc + d];
    float xv  = bf2f(xi[t * DIc + d]);
    float dtx = dtv * xv;
    const float* bc = xdb + t * XPN;
    float accv = 0.0f;
#pragma unroll
    for (int s = 0; s < DSc; ++s) {
      float dA = __expf(dtv * Ar[s]);
      hs[s] = dA * hs[s] + dtx * bc[DTRc + s];
      accv += hs[s] * bc[DTRc + DSc + s];
    }
    y[t * DIc + d] = accv + xv * Dd;
  }
}

// ---------------- gating: yg = y * silu(z) -------------------------------
__global__ void gate_kernel(const float* __restrict__ y,
                            const ushort_t* __restrict__ xz,
                            ushort_t* __restrict__ yg) {
  size_t total = (size_t)Mrows * DIc;
  for (size_t idx = blockIdx.x * (size_t)blockDim.x + threadIdx.x; idx < total;
       idx += (size_t)gridDim.x * blockDim.x) {
    int d = (int)(idx % DIc);
    size_t t = idx / DIc;
    float z = bf2f(xz[t * (2 * DIc) + DIc + d]);
    float v = y[idx] * (z / (1.0f + __expf(-z)));
    yg[idx] = f2bf(v);
  }
}

// ---------------- residual (fwd + flip(bwd) + h) + LayerNorm -> bf16 -----
__global__ void resid_ln_kernel(const float* __restrict__ fwd,
                                const float* __restrict__ bwd,
                                const float* __restrict__ hf,
                                const float* __restrict__ g,
                                const float* __restrict__ bb,
                                ushort_t* __restrict__ out) {
  int t = blockIdx.x;                  // token index 0..Mrows-1
  int b = t / Lc, l = t % Lc;
  const float* frow = fwd + (size_t)t * DIMc;
  const float* brow = bwd + ((size_t)b * Lc + (Lc - 1 - l)) * DIMc;
  const float* hrow = hf  + (size_t)t * DIMc;
  float vals[8];
  float s1 = 0.f, s2 = 0.f;
#pragma unroll
  for (int i = 0; i < 8; ++i) {
    int j = threadIdx.x + i * 256;
    float v = frow[j] + brow[j] + hrow[j];
    vals[i] = v; s1 += v; s2 += v * v;
  }
  __shared__ float r1[256], r2[256];
  r1[threadIdx.x] = s1; r2[threadIdx.x] = s2;
  __syncthreads();
  for (int st = 128; st > 0; st >>= 1) {
    if (threadIdx.x < st) {
      r1[threadIdx.x] += r1[threadIdx.x + st];
      r2[threadIdx.x] += r2[threadIdx.x + st];
    }
    __syncthreads();
  }
  float mu  = r1[0] * (1.0f / DIMc);
  float var = r2[0] * (1.0f / DIMc) - mu * mu;
  float rs  = rsqrtf(var + 1e-5f);
#pragma unroll
  for (int i = 0; i < 8; ++i) {
    int j = threadIdx.x + i * 256;
    out[(size_t)t * DIMc + j] = f2bf((vals[i] - mu) * rs * g[j] + bb[j]);
  }
}

// ---------------- host orchestration ----------------
extern "C" void kernel_launch(void* const* d_in, const int* in_sizes, int n_in,
                              void* d_out, int out_size, void* d_ws,
                              size_t ws_size, hipStream_t stream) {
  (void)in_sizes; (void)n_in; (void)out_size; (void)ws_size;

  const float* img   = (const float*)d_in[0];
  const float* w_in  = (const float*)d_in[1];
  const float* b_in  = (const float*)d_in[2];
  const float* ln_g  = (const float*)d_in[3];
  const float* ln_b  = (const float*)d_in[4];
  const float* w_out = (const float*)d_in[5];
  const float* b_out = (const float*)d_in[6];
  // mp[m]: 0=in_w 1=conv_w 2=conv_b 3=xproj_w 4=dt_w 5=dt_b 6=A_log 7=D 8=out_w
  const float* mp[2][9];
  for (int m = 0; m < 2; ++m)
    for (int i = 0; i < 9; ++i)
      mp[m][i] = (const float*)d_in[7 + m * 9 + i];

  char* base = (char*)d_ws;
  size_t off = 0;
  auto alloc = [&](size_t bytes) -> void* {
    void* p = base + off;
    off = (off + bytes + 255) & ~(size_t)255;
    return p;
  };

  ushort_t* imgbf    = (ushort_t*)alloc((size_t)Mrows * FUSEDc * 2);
  ushort_t* w_in_bf  = (ushort_t*)alloc((size_t)DIMc * FUSEDc * 2);
  ushort_t* w_out_bf = (ushort_t*)alloc((size_t)DIMc * DIMc * 2);
  ushort_t *in_w_bf[2], *xproj_bf[2], *dtw_bf[2], *outw_bf[2];
  for (int m = 0; m < 2; ++m) {
    in_w_bf[m]  = (ushort_t*)alloc((size_t)2 * DIc * DIMc * 2);
    xproj_bf[m] = (ushort_t*)alloc((size_t)XPN * DIc * 2);
    dtw_bf[m]   = (ushort_t*)alloc((size_t)DIc * DTRc * 2);
    outw_bf[m]  = (ushort_t*)alloc((size_t)DIMc * DIc * 2);
  }
  float*    hfp     = (float*)alloc((size_t)Mrows * DIMc * 4);
  ushort_t* hbf     = (ushort_t*)alloc((size_t)Mrows * DIMc * 2);
  ushort_t* hflipbf = (ushort_t*)alloc((size_t)Mrows * DIMc * 2);
  ushort_t* xzbf    = (ushort_t*)alloc((size_t)Mrows * 2 * DIc * 2);
  ushort_t* xibf    = (ushort_t*)alloc((size_t)Mrows * DIc * 2);
  float*    xdbf    = (float*)alloc((size_t)Mrows * XPN * 4);
  ushort_t* xdbbf   = (ushort_t*)alloc((size_t)Mrows * XPN * 2);
  float*    dtf     = (float*)alloc((size_t)Mrows * DIc * 4);
  float*    yf      = (float*)alloc((size_t)Mrows * DIc * 4);
  ushort_t* ygbf    = (ushort_t*)alloc((size_t)Mrows * DIc * 2);
  float*    dir_out[2];
  dir_out[0] = (float*)alloc((size_t)Mrows * DIMc * 4);
  dir_out[1] = (float*)alloc((size_t)Mrows * DIMc * 4);
  ushort_t* rbf = (ushort_t*)alloc((size_t)Mrows * DIMc * 2);

  auto nblk = [](size_t n) { return (unsigned)((n + 255) / 256); };
  auto cvt = [&](const float* s, ushort_t* d, size_t n) {
    f32_to_bf16_kernel<<<nblk(n), 256, 0, stream>>>(s, d, n);
  };
  auto gemm = [&](const ushort_t* Aa, int lda, const ushort_t* Ww, int ldb,
                  const float* bias, float* oF, ushort_t* oH, int ldo, int N,
                  int K, int act) {
    dim3 grid((N + 127) / 128, Mrows / 128);
    gemm_bf16_wmma<<<grid, 256, 0, stream>>>(Aa, lda, Ww, ldb, bias, oF, oH,
                                             ldo, Mrows, N, K, act);
  };

  // ---- down-convert activations & all GEMM weights to bf16 ----
  cvt(img,   imgbf,    (size_t)Mrows * FUSEDc);
  cvt(w_in,  w_in_bf,  (size_t)DIMc * FUSEDc);
  cvt(w_out, w_out_bf, (size_t)DIMc * DIMc);
  for (int m = 0; m < 2; ++m) {
    cvt(mp[m][0], in_w_bf[m],  (size_t)2 * DIc * DIMc);
    cvt(mp[m][3], xproj_bf[m], (size_t)XPN * DIc);
    cvt(mp[m][4], dtw_bf[m],   (size_t)DIc * DTRc);
    cvt(mp[m][8], outw_bf[m],  (size_t)DIMc * DIc);
  }

  // ---- h = gelu(img @ w_in^T + b_in) -> f32 (residual) + bf16 (GEMM A) ----
  gemm(imgbf, FUSEDc, w_in_bf, FUSEDc, b_in, hfp, hbf, DIMc, DIMc, FUSEDc, 1);
  flip_bf16_kernel<<<nblk((size_t)Mrows * DIMc), 256, 0, stream>>>(hbf, hflipbf);

  // ---- forward (m=0) and backward (m=1) mamba branches ----
  for (int m = 0; m < 2; ++m) {
    const ushort_t* hsrc = (m == 0) ? hbf : hflipbf;
    // xz = h @ in_w^T  (N=8192, K=2048) -> bf16
    gemm(hsrc, DIMc, in_w_bf[m], DIMc, nullptr, nullptr, xzbf, 2 * DIc,
         2 * DIc, DIMc, 0);
    // xi = silu(depthwise_conv(xz[:, :DI]))
    conv_silu_kernel<<<nblk((size_t)Mrows * DIc), 256, 0, stream>>>(
        xzbf, mp[m][1], mp[m][2], xibf);
    // xdb = xi @ xproj^T  (N=160, K=4096) -> f32 (scan B/C) + bf16 (dtr)
    gemm(xibf, DIc, xproj_bf[m], DIc, nullptr, xdbf, xdbbf, XPN, XPN, DIc, 0);
    // dt = softplus(dtr @ dt_w^T + dt_b)  (A = xdb cols 0..127, lda=160)
    gemm(xdbbf, XPN, dtw_bf[m], DTRc, mp[m][5], dtf, nullptr, DIc, DIc, DTRc, 2);
    // selective scan (sequential in L)
    scan_kernel<<<(Bc * DIc + 255) / 256, 256, 0, stream>>>(
        xibf, dtf, xdbf, mp[m][6], mp[m][7], yf);
    // gate: yg = y * silu(z)
    gate_kernel<<<nblk((size_t)Mrows * DIc), 256, 0, stream>>>(yf, xzbf, ygbf);
    // branch out: yg @ out_w^T  (N=2048, K=4096) -> f32
    gemm(ygbf, DIc, outw_bf[m], DIc, nullptr, dir_out[m], nullptr, DIMc, DIMc,
         DIc, 0);
  }

  // ---- r = LN(fwd + flip(bwd) + h) -> bf16 ----
  resid_ln_kernel<<<Mrows, 256, 0, stream>>>(dir_out[0], dir_out[1], hfp, ln_g,
                                             ln_b, rbf);
  // ---- out = r @ w_out^T + b_out -> d_out (f32) ----
  gemm(rbf, DIMc, w_out_bf, DIMc, b_out, (float*)d_out, nullptr, DIMc, DIMc,
       DIMc, 0);
}
